// FastVolterra1_88184268521672
// MI455X (gfx1250) — compile-verified, matrix operand order checked by reading, not run on hardware
//
#include <hip/hip_runtime.h>

// ---------------------------------------------------------------------------
// y = irfft(rfft(x) * softmax(w1))  ==  x (circ-conv) h,  h = irfft(softmax(w1))
// => GEMM  Y[32768x1024] = X[32768x1024] * C[1024x1024],  C[k][n] = h[(n-k)&1023]
// Split-bf16 (hi+lo) GEMM on v_wmma_f32_16x16x32_bf16, 3 products per K-step.
// A tile staged raw-f32 via GLOBAL_LOAD_ASYNC_TO_LDS (ASYNCcnt), double-buffered,
// one barrier per K-step; bf16 split done in-register during fragment build.
// ---------------------------------------------------------------------------

typedef __attribute__((ext_vector_type(16))) __bf16 v16bf;
typedef __attribute__((ext_vector_type(8)))  float  v8f;

#define DDIM 1024
#define ASTRIDE 36   // f32 per LDS row: 32 data + 4 pad (144B, 16B-aligned rows)

static __device__ __forceinline__ unsigned int pack2(__bf16 a, __bf16 b) {
    union { __bf16 h[2]; unsigned int u; } q;
    q.h[0] = a; q.h[1] = b;
    return q.u;
}

// ---- Kernel 1: gate = softmax(w1); h = irfft(gate) -------------------------
__global__ void __launch_bounds__(1024)
prep_h_kernel(const float* __restrict__ w1, float* __restrict__ h) {
    __shared__ float red[1024];
    __shared__ float g[513];
    const int t = threadIdx.x;

    float v = (t < 513) ? w1[t] : -3.4e38f;
    red[t] = v; __syncthreads();
    for (int s = 512; s > 0; s >>= 1) {
        if (t < s) red[t] = fmaxf(red[t], red[t + s]);
        __syncthreads();
    }
    const float mx = red[0];
    __syncthreads();

    float e = (t < 513) ? __expf(w1[t] - mx) : 0.0f;
    red[t] = e; __syncthreads();
    for (int s = 512; s > 0; s >>= 1) {
        if (t < s) red[t] += red[t + s];
        __syncthreads();
    }
    const float inv = 1.0f / red[0];
    if (t < 513) g[t] = e * inv;
    __syncthreads();

    // h[n] = (1/D) * ( g0 + (-1)^n gD/2 + 2*sum_{k=1}^{D/2-1} gk cos(2*pi*k*n/D) )
    const int n = t;
    float s = g[0] + ((n & 1) ? -g[512] : g[512]);
    const float w = 6.283185307179586f / 1024.0f;
    for (int k = 1; k < 512; ++k) {
        int ph = (k * n) & 1023;                 // exact integer phase reduction
        s += 2.0f * g[k] * __cosf((float)ph * w);
    }
    h[n] = s * (1.0f / 1024.0f);
}

// ---- Kernel 2: build bf16 hi/lo circulant in WMMA B-fragment order --------
// Fragment = 32(K) x 16(N).  Assumed gfx1250 16-bit B layout (mirror of the
// documented A 16x32 layout, N striped across lanes):
//   lane   = n' + 16*((k'>>3)&1)     elem j = (k'&7) + 8*(k'>>4)
// Fragment f = kt*64 + nTile stored at f*512, lane-major (lane*16 + j):
// each lane's 16 bf16 form one contiguous 32-byte chunk.
__global__ void __launch_bounds__(256)
build_B_kernel(const float* __restrict__ h,
               unsigned short* __restrict__ Bhi,
               unsigned short* __restrict__ Blo) {
    const int idx = blockIdx.x * 256 + threadIdx.x;   // 0 .. 1048575
    const int k = idx >> 10;
    const int n = idx & 1023;
    const float val = h[(n - k) & 1023];

    const __bf16 hi = (__bf16)val;
    const __bf16 lo = (__bf16)(val - (float)hi);

    const int kt = k >> 5, kp = k & 31;
    const int nT = n >> 4, np = n & 15;
    const int laneHi = (kp >> 3) & 1;
    const int lane   = np + (laneHi << 4);
    const int j      = (kp & 7) + ((kp >> 4) << 3);
    const size_t off = ((size_t)(kt * 64 + nT) << 9) + (size_t)lane * 16 + j;

    ((__bf16*)Bhi)[off] = hi;
    ((__bf16*)Blo)[off] = lo;
}

// ---- Kernel 3: split-bf16 WMMA GEMM ---------------------------------------
// 256 threads = 8 waves.  WG tile 128(M) x 128(N).  Wave (wm 0..3, wn 0..1)
// owns 2x4 16x16 output tiles -> 24 WMMAs per wave per K-step.
// A tile (128x32 f32) double-buffered in LDS via async-to-LDS DMA.
__global__ void __launch_bounds__(256)
volterra_gemm_kernel(const float* __restrict__ X,
                     const unsigned short* __restrict__ Bhi,
                     const unsigned short* __restrict__ Blo,
                     float* __restrict__ Y) {
    __shared__ float ldsA[2][128 * ASTRIDE];

    const int tid    = threadIdx.x;
    const int lane   = tid & 31;
    const int wave   = tid >> 5;
    const int wm     = wave >> 1;          // 0..3
    const int wn     = wave & 1;           // 0..1
    const int mBase  = blockIdx.y * 128;
    const int nBase  = blockIdx.x * 128;
    const int m_lane = lane & 15;
    const int laneHi = lane >> 4;

    const v8f zero = {0.f, 0.f, 0.f, 0.f, 0.f, 0.f, 0.f, 0.f};
    v8f acc[2][4];
    #pragma unroll
    for (int mi = 0; mi < 2; ++mi)
        #pragma unroll
        for (int ni = 0; ni < 4; ++ni) acc[mi][ni] = zero;

    // Issue the 4 async 16B chunks of this thread's share of A tile `kt`
    // into LDS buffer `buf` (128 rows x 32 f32, row stride ASTRIDE).
    auto stage = [&](int kt, int buf) {
        const int kc = kt * 32;
        #pragma unroll
        for (int i = 0; i < 4; ++i) {
            const int linear = tid + i * 256;       // float4 units, 0..1023
            const int row  = linear >> 3;           // 0..127
            const int quad = linear & 7;            // 0..7
            const unsigned long long ga =
                (unsigned long long)(X + (size_t)(mBase + row) * DDIM + kc + quad * 4);
            const unsigned int la =
                (unsigned int)(size_t)&ldsA[buf][row * ASTRIDE + quad * 4];
            asm volatile("global_load_async_to_lds_b128 %0, %1, off"
                         :: "v"(la), "v"(ga) : "memory");
        }
    };

    stage(0, 0);

    for (int kt = 0; kt < 32; ++kt) {
        // my DMA chunks done -> barrier -> everyone's chunks visible
        asm volatile("s_wait_asynccnt 0x0" ::: "memory");
        __syncthreads();
        if (kt + 1 < 32) stage(kt + 1, (kt + 1) & 1);

        const float* Abuf = ldsA[kt & 1];

        // ---- A fragments: read f32 pairs from LDS, split to bf16 hi/lo ----
        v16bf aHi[2], aLo[2];
        #pragma unroll
        for (int mi = 0; mi < 2; ++mi) {
            const int row = wm * 32 + mi * 16 + m_lane;
            union { v16bf v; unsigned int u[8]; } UH, UL;
            #pragma unroll
            for (int p = 0; p < 8; ++p) {
                // K' pair base: p<4 -> 2p + 8*laneHi ; p>=4 -> 16 + 2(p-4) + 8*laneHi
                const int kp = ((p & 3) << 1) + (laneHi << 3) + ((p >> 2) << 4);
                const float2 f = *(const float2*)&Abuf[row * ASTRIDE + kp];
                const __bf16 h0 = (__bf16)f.x, h1 = (__bf16)f.y;
                const __bf16 l0 = (__bf16)(f.x - (float)h0);
                const __bf16 l1 = (__bf16)(f.y - (float)h1);
                UH.u[p] = pack2(h0, h1);
                UL.u[p] = pack2(l0, l1);
            }
            aHi[mi] = UH.v;
            aLo[mi] = UL.v;
        }

        // ---- B fragments: contiguous 32B per lane, straight from L2 ----
        v16bf bHi[4], bLo[4];
        #pragma unroll
        for (int ni = 0; ni < 4; ++ni) {
            const int nTile = (nBase >> 4) + wn * 4 + ni;
            const size_t off = ((size_t)(kt * 64 + nTile) << 9) + (size_t)lane * 16;
            bHi[ni] = *(const v16bf*)((const __bf16*)Bhi + off);
            bLo[ni] = *(const v16bf*)((const __bf16*)Blo + off);
        }

        // ---- 24 WMMAs: AhiBhi + AloBhi + AhiBlo for 2x4 tiles ----
        #pragma unroll
        for (int mi = 0; mi < 2; ++mi)
            #pragma unroll
            for (int ni = 0; ni < 4; ++ni) {
                acc[mi][ni] = __builtin_amdgcn_wmma_f32_16x16x32_bf16(
                    false, aHi[mi], false, bHi[ni], (short)0, acc[mi][ni], false, false);
                acc[mi][ni] = __builtin_amdgcn_wmma_f32_16x16x32_bf16(
                    false, aLo[mi], false, bHi[ni], (short)0, acc[mi][ni], false, false);
                acc[mi][ni] = __builtin_amdgcn_wmma_f32_16x16x32_bf16(
                    false, aHi[mi], false, bLo[ni], (short)0, acc[mi][ni], false, false);
            }
    }

    // ---- store: C/D layout M = r + 8*laneHi, N = lane&15 ----
    #pragma unroll
    for (int mi = 0; mi < 2; ++mi)
        #pragma unroll
        for (int ni = 0; ni < 4; ++ni) {
            const int rbase = mBase + wm * 32 + mi * 16 + 8 * laneHi;
            const int col   = nBase + wn * 64 + ni * 16 + m_lane;
            #pragma unroll
            for (int r = 0; r < 8; ++r)
                Y[(size_t)(rbase + r) * DDIM + col] = acc[mi][ni][r];
        }
}

// ---------------------------------------------------------------------------
extern "C" void kernel_launch(void* const* d_in, const int* in_sizes, int n_in,
                              void* d_out, int out_size, void* d_ws, size_t ws_size,
                              hipStream_t stream) {
    const float* hidden = (const float*)d_in[0];   // [B*L, 1024] f32
    const float* w1     = (const float*)d_in[1];   // [513] f32
    float*       out    = (float*)d_out;           // [B*L, 1024] f32
    const int rows = in_sizes[0] / DDIM;           // 32768

    char* ws = (char*)d_ws;
    float*          h   = (float*)ws;                                  // 4 KB
    unsigned short* Bhi = (unsigned short*)(ws + 4096);                // 2 MB
    unsigned short* Blo = (unsigned short*)(ws + 4096 + (1u << 21));   // 2 MB

    prep_h_kernel<<<1, 1024, 0, stream>>>(w1, h);
    build_B_kernel<<<4096, 256, 0, stream>>>(h, Bhi, Blo);

    dim3 grid(DDIM / 128, rows / 128);             // (8, 256)
    volterra_gemm_kernel<<<grid, 256, 0, stream>>>(hidden, Bhi, Blo, out);
}
